// MultiHeadAttention_44186623542094
// MI455X (gfx1250) — matrix operand order repaired and checked by reading, not compiled, not run on hardware
//
#include <hip/hip_runtime.h>
#include <hip/hip_bf16.h>

typedef __attribute__((ext_vector_type(16))) _Float16 v16h;
typedef __attribute__((ext_vector_type(8)))  _Float16 v8h;
typedef __attribute__((ext_vector_type(8)))  float    v8f;

// Problem constants (reference: B=4, N=2048, D=1024, H=8, HD=128)
constexpr int Bc  = 4;
constexpr int Nc  = 2048;
constexpr int Dc  = 1024;
constexpr int Hc  = 8;
constexpr int HDc = 128;
constexpr int Mc  = Bc * Nc;   // 8192 rows of x
constexpr int Kc  = Dc;        // reduction dim for projections

#define WMMA_F16(A, B, C) \
  __builtin_amdgcn_wmma_f32_16x16x32_f16(false, (A), false, (B), (short)0, (C), false, false)

static __device__ __forceinline__ v16h cat8(v8h lo, v8h hi) {
  v16h r;
#pragma unroll
  for (int i = 0; i < 8; ++i) { r[i] = lo[i]; r[8 + i] = hi[i]; }
  return r;
}

// ---------------------------------------------------------------------------
// fp32 -> fp16 conversion
// ---------------------------------------------------------------------------
__global__ void cvt_f32_f16(const float* __restrict__ src,
                            _Float16* __restrict__ dst, int n) {
  int i = blockIdx.x * blockDim.x + threadIdx.x;
  if (i < n) dst[i] = (_Float16)src[i];
}

// ---------------------------------------------------------------------------
// GEMM: out[m,n] = sum_k A[m,k] * W[n,k]   (A @ W^T), A:[Mc,Kc] W:[Dc,Kc]
// MODE 0: store f16 as [B,H,N,HD]   (Q, K)
// MODE 1: store f16 as [B,H,HD,N]   (V transposed)
// MODE 2: store f32 as [Mc,Dc]      (final output)
// Block: 128 threads = 4 waves; wave computes 32x64 (8 acc, B reused x2);
// block covers 128 rows x 64 cols.
// ---------------------------------------------------------------------------
template <int MODE>
__global__ __launch_bounds__(128) void gemm16(const _Float16* __restrict__ A,
                                              const _Float16* __restrict__ W,
                                              void* __restrict__ outp) {
  const int lane = threadIdx.x & 31;
  const int wv   = threadIdx.x >> 5;
  const int col  = lane & 15;
  const int half = lane >> 4;
  const int n0   = blockIdx.x * 64;
  const int m0   = blockIdx.y * 128 + wv * 32;

  v8f acc[2][4] = {};
  const _Float16* arow0 = A + (size_t)(m0 + col) * Kc;
  const _Float16* arow1 = A + (size_t)(m0 + 16 + col) * Kc;

  for (int k0 = 0; k0 < Kc; k0 += 32) {
    // A fragments: lane holds row (col); K chunks [8*half,+8) and [16+8*half,+8)
    v16h a0 = cat8(*reinterpret_cast<const v8h*>(arow0 + k0 + 8 * half),
                   *reinterpret_cast<const v8h*>(arow0 + k0 + 16 + 8 * half));
    v16h a1 = cat8(*reinterpret_cast<const v8h*>(arow1 + k0 + 8 * half),
                   *reinterpret_cast<const v8h*>(arow1 + k0 + 16 + 8 * half));
    if (k0 + 32 < Kc) {
      __builtin_prefetch(arow0 + k0 + 32, 0, 1);
      __builtin_prefetch(arow1 + k0 + 32, 0, 1);
    }
#pragma unroll
    for (int t = 0; t < 4; ++t) {
      // B fragment: B[k,n] = W[n,k]; lane = column n, K chunks [8*half,+8) and [16+8*half,+8)
      const _Float16* wp = W + (size_t)(n0 + 16 * t + col) * Kc + k0 + 8 * half;
      v16h b = cat8(*reinterpret_cast<const v8h*>(wp),
                    *reinterpret_cast<const v8h*>(wp + 16));
      acc[0][t] = WMMA_F16(a0, b, acc[0][t]);
      acc[1][t] = WMMA_F16(a1, b, acc[1][t]);
    }
  }

#pragma unroll
  for (int mi = 0; mi < 2; ++mi) {
#pragma unroll
    for (int t = 0; t < 4; ++t) {
#pragma unroll
      for (int r = 0; r < 8; ++r) {
        const int Mr = m0 + 16 * mi + r + 8 * half;  // output row (token index)
        const int Nn = n0 + 16 * t + col;            // output col (model dim)
        const float v = acc[mi][t][r];
        if (MODE == 2) {
          ((float*)outp)[(size_t)Mr * Dc + Nn] = v;
        } else {
          const int bb = Mr >> 11, tok = Mr & (Nc - 1);
          const int hh = Nn >> 7,  dd  = Nn & (HDc - 1);
          _Float16* o = (_Float16*)outp;
          if (MODE == 0)
            o[(((size_t)bb * Hc + hh) * Nc + tok) * HDc + dd] = (_Float16)v;
          else
            o[(((size_t)bb * Hc + hh) * HDc + dd) * Nc + tok] = (_Float16)v;
        }
      }
    }
  }
}

// ---------------------------------------------------------------------------
// Flash attention, causal. Block = 4 waves = 64 consecutive queries of one
// (b,h); K/V tiles staged in LDS once per 32-key block and shared by all
// waves. All waves run to the block diagonal so __syncthreads is uniform;
// fully-masked iterations contribute exp(-huge)=0 with alpha=1 (safe).
// Q,K: [B,H,N,HD] f16.  Vt: [B,H,HD,N] f16.  ctx out: [B,N,D] f16.
// ---------------------------------------------------------------------------
__global__ __launch_bounds__(128) void attn_fwd(const _Float16* __restrict__ Q,
                                                const _Float16* __restrict__ Km,
                                                const _Float16* __restrict__ Vt,
                                                _Float16* __restrict__ ctx) {
  __shared__ __align__(16) _Float16 Kl[32][136];      // 32 keys x 128 dims (+8 pad)
  __shared__ __align__(16) _Float16 Vl[128][40];      // 128 dims x 32 keys (+8 pad)
  __shared__ __align__(16) _Float16 pbuf[4][16][40];  // per-wave P staging

  const int tid  = threadIdx.x;
  const int lane = tid & 31;
  const int wv   = tid >> 5;
  const int col  = lane & 15;
  const int half = lane >> 4;
  const int bh   = blockIdx.y;            // 0..31 = b*H + h
  const int mblk = blockIdx.x * 64;       // block's query base
  const int m0   = mblk + wv * 16;        // this wave's query tile

  const size_t qkoff = (size_t)bh * Nc * HDc;
  const _Float16* Qb = Q  + qkoff;
  const _Float16* Kb = Km + qkoff;
  const _Float16* Vb = Vt + (size_t)bh * HDc * Nc;
  const float scale = 0.08838834764831845f;  // 1/sqrt(128)

  // Preload Q fragments: 4 K-chunks of 32 covering HD=128
  v16h q[4];
  {
    const _Float16* qr = Qb + (size_t)(m0 + col) * HDc;
#pragma unroll
    for (int kc = 0; kc < 4; ++kc)
      q[kc] = cat8(*reinterpret_cast<const v8h*>(qr + 32 * kc + 8 * half),
                   *reinterpret_cast<const v8h*>(qr + 32 * kc + 16 + 8 * half));
  }

  v8f o[8] = {};
  float mrow[8], lrow[8];
#pragma unroll
  for (int r = 0; r < 8; ++r) { mrow[r] = -3.0e38f; lrow[r] = 0.0f; }

  const int jend = mblk + 63;                 // block-uniform causal bound
  for (int j0 = 0; j0 <= jend; j0 += 32) {
    // ---- cooperative staging: K tile 32x128 (8KB), V tile 128x32 (8KB)
    {
      const int krow = tid >> 2, kseg = (tid & 3) * 32;       // 64B of a K row
      const _Float16* ks = Kb + (size_t)(j0 + krow) * HDc + kseg;
#pragma unroll
      for (int c = 0; c < 4; ++c)
        *reinterpret_cast<v8h*>(&Kl[krow][kseg + 8 * c]) =
            *reinterpret_cast<const v8h*>(ks + 8 * c);
      const _Float16* vs = Vb + (size_t)tid * Nc + j0;        // 64B of a Vt row
#pragma unroll
      for (int c = 0; c < 4; ++c)
        *reinterpret_cast<v8h*>(&Vl[tid][8 * c]) =
            *reinterpret_cast<const v8h*>(vs + 8 * c);
    }
    __syncthreads();

    // ---- S = Q @ K^T for 2 sub-tiles of 16 keys
    v8f s[2];
#pragma unroll
    for (int sub = 0; sub < 2; ++sub) {
      v8f acc = {};
      const _Float16* kr = &Kl[16 * sub + col][8 * half];
#pragma unroll
      for (int kc = 0; kc < 4; ++kc) {
        v16h b = cat8(*reinterpret_cast<const v8h*>(kr + 32 * kc),
                      *reinterpret_cast<const v8h*>(kr + 32 * kc + 16));
        acc = WMMA_F16(q[kc], b, acc);
      }
      s[sub] = acc;
    }

    // ---- scale + causal mask; per-row running max
    float mx[8];
#pragma unroll
    for (int r = 0; r < 8; ++r) {
      const int mq = m0 + r + 8 * half;  // query index
      float v0 = (j0 + col      <= mq) ? s[0][r] * scale : -3.0e38f;
      float v1 = (j0 + 16 + col <= mq) ? s[1][r] * scale : -3.0e38f;
      s[0][r] = v0; s[1][r] = v1;
      mx[r] = fmaxf(v0, v1);
    }
#pragma unroll
    for (int d = 1; d < 16; d <<= 1) {
#pragma unroll
      for (int r = 0; r < 8; ++r)
        mx[r] = fmaxf(mx[r], __shfl_xor(mx[r], d, 32));
    }

    // ---- online softmax: P = exp(S - m_new); stage P into LDS (C -> A layout)
    float alpha[8], ps[8];
#pragma unroll
    for (int r = 0; r < 8; ++r) {
      const float mn = fmaxf(mrow[r], mx[r]);
      alpha[r] = __expf(mrow[r] - mn);
      mrow[r]  = mn;
      const float p0 = __expf(s[0][r] - mn);
      const float p1 = __expf(s[1][r] - mn);
      ps[r] = p0 + p1;
      pbuf[wv][r + 8 * half][col]      = (_Float16)p0;
      pbuf[wv][r + 8 * half][16 + col] = (_Float16)p1;
    }
#pragma unroll
    for (int d = 1; d < 16; d <<= 1) {
#pragma unroll
      for (int r = 0; r < 8; ++r)
        ps[r] += __shfl_xor(ps[r], d, 32);
    }
#pragma unroll
    for (int r = 0; r < 8; ++r) lrow[r] = lrow[r] * alpha[r] + ps[r];

    // same-wave DS RAW hazard: P stores must land before A-layout reads
    asm volatile("s_wait_dscnt 0" ::: "memory");

    // ---- P in A layout: lane = query row (col), K chunks [8*half,+8) and [16+8*half,+8)
    v16h P;
    {
      const _Float16* pr = &pbuf[wv][col][0];
      P = cat8(*reinterpret_cast<const v8h*>(pr + 8 * half),
               *reinterpret_cast<const v8h*>(pr + 16 + 8 * half));
    }

    // ---- O = O*alpha + P @ V  over 8 head-dim tiles of 16
#pragma unroll
    for (int nt = 0; nt < 8; ++nt) {
#pragma unroll
      for (int r = 0; r < 8; ++r) o[nt][r] *= alpha[r];
      const _Float16* vr = &Vl[16 * nt + col][8 * half];
      v16h bv = cat8(*reinterpret_cast<const v8h*>(vr),
                     *reinterpret_cast<const v8h*>(vr + 16));
      o[nt] = WMMA_F16(P, bv, o[nt]);
    }
    __syncthreads();  // K/V tiles consumed before next staging overwrites
  }

  // ---- normalize and write ctx[b, tok, h*HD + d] (f16)
  const int bb = bh >> 3, hh = bh & 7;
#pragma unroll
  for (int nt = 0; nt < 8; ++nt) {
#pragma unroll
    for (int r = 0; r < 8; ++r) {
      const float val = o[nt][r] / lrow[r];
      ctx[((size_t)bb * Nc + m0 + r + 8 * half) * Dc + hh * HDc + 16 * nt + col] =
          (_Float16)val;
    }
  }
}

// ---------------------------------------------------------------------------
// Launch
// ---------------------------------------------------------------------------
extern "C" void kernel_launch(void* const* d_in, const int* in_sizes, int n_in,
                              void* d_out, int out_size, void* d_ws, size_t ws_size,
                              hipStream_t stream) {
  (void)in_sizes; (void)n_in; (void)out_size; (void)ws_size;
  const float* x  = (const float*)d_in[0];
  const float* Wq = (const float*)d_in[1];
  const float* Wk = (const float*)d_in[2];
  const float* Wv = (const float*)d_in[3];
  const float* Wo = (const float*)d_in[4];
  float* out = (float*)d_out;

  const size_t MK = (size_t)Mc * Kc;   // 8,388,608  (x / q / k / v / ctx elems)
  const size_t WK = (size_t)Dc * Kc;   // 1,048,576  (weight elems)

  _Float16* xh  = (_Float16*)d_ws;
  _Float16* wqh = xh  + MK;
  _Float16* wkh = wqh + WK;
  _Float16* wvh = wkh + WK;
  _Float16* woh = wvh + WK;
  _Float16* qb  = woh + WK;
  _Float16* kb  = qb  + MK;
  _Float16* vtb = kb  + MK;
  _Float16* ctx = xh;                  // x no longer needed after V projection

  const int thr = 256;
  cvt_f32_f16<<<(int)(MK / thr), thr, 0, stream>>>(x,  xh,  (int)MK);
  cvt_f32_f16<<<(int)(WK / thr), thr, 0, stream>>>(Wq, wqh, (int)WK);
  cvt_f32_f16<<<(int)(WK / thr), thr, 0, stream>>>(Wk, wkh, (int)WK);
  cvt_f32_f16<<<(int)(WK / thr), thr, 0, stream>>>(Wv, wvh, (int)WK);
  cvt_f32_f16<<<(int)(WK / thr), thr, 0, stream>>>(Wo, woh, (int)WK);

  dim3 gg(Dc / 64, Mc / 128);          // (16, 64)
  gemm16<0><<<gg, 128, 0, stream>>>(xh, wqh, qb);   // Q   [B,H,N,HD]
  gemm16<0><<<gg, 128, 0, stream>>>(xh, wkh, kb);   // K   [B,H,N,HD]
  gemm16<1><<<gg, 128, 0, stream>>>(xh, wvh, vtb);  // V^T [B,H,HD,N]

  attn_fwd<<<dim3(Nc / 64, Bc * Hc), 128, 0, stream>>>(qb, kb, vtb, ctx);

  gemm16<2><<<gg, 128, 0, stream>>>(ctx, woh, out); // ctx @ Wo^T -> f32
}
